// MultiHeadAttention_7198365188622
// MI455X (gfx1250) — compile-verified, matrix operand order checked by reading, not run on hardware
//
#include <hip/hip_runtime.h>
#include <hip/hip_bf16.h>

// ---------------------------------------------------------------------------
// Problem constants: B=4, L=2048, D=1024, H=16, DK=DV=64
// ---------------------------------------------------------------------------
constexpr int BB  = 4;
constexpr int LL  = 2048;
constexpr int DD  = 1024;     // = H*DK = H*DV
constexpr int HH  = 16;
constexpr int DKK = 64;

typedef __bf16 v16bf __attribute__((ext_vector_type(16)));
typedef float  v8f   __attribute__((ext_vector_type(8)));
typedef unsigned int v4u  __attribute__((ext_vector_type(4)));
typedef int          v4i_ __attribute__((ext_vector_type(4)));
typedef int          v8i_ __attribute__((ext_vector_type(8)));

__device__ __forceinline__ v8f vzero() {
    v8f x = {0.f, 0.f, 0.f, 0.f, 0.f, 0.f, 0.f, 0.f};
    return x;
}

__device__ __forceinline__ v8f wmma_bf16(v16bf a, v16bf b, v8f c) {
    return __builtin_amdgcn_wmma_f32_16x16x32_bf16(
        false, a, false, b, (short)0, c, false, false);
}

// K index of VGPR-half v (0..7) per ISA 7.12.2 "16-bit A-Matrix 16x32" table.
__device__ __forceinline__ int wmma_k_base(int v, int lane) {
    return 16 * (v >> 2) + 2 * (v & 3) + 8 * ((lane >> 4) & 1);
}

// A fragment: element (m,k) = a[m*lda + k]; lane holds row m = lane&15.
__device__ __forceinline__ v16bf load_a_frag(const __bf16* a, int lda, int lane) {
    const int m = lane & 15;
    v16bf f;
#pragma unroll
    for (int v = 0; v < 8; ++v) {
        const int k = wmma_k_base(v, lane);
        f[2 * v]     = a[(size_t)m * lda + k];
        f[2 * v + 1] = a[(size_t)m * lda + k + 1];
    }
    return f;
}

// B fragment: element (k,n) = w[k*ldw + n]; lane holds column n = lane&15.
__device__ __forceinline__ v16bf load_b_frag(const __bf16* w, int ldw, int lane) {
    const int n = lane & 15;
    v16bf f;
#pragma unroll
    for (int v = 0; v < 8; ++v) {
        const int k = wmma_k_base(v, lane);
        f[2 * v]     = w[(size_t)k * ldw + n];
        f[2 * v + 1] = w[(size_t)(k + 1) * ldw + n];
    }
    return f;
}

// B fragment of a transposed row-major matrix: element (k,n) = mrow[n*ldm + k].
__device__ __forceinline__ v16bf load_bt_frag(const __bf16* mrow, int ldm, int lane) {
    const int n = lane & 15;
    v16bf f;
#pragma unroll
    for (int v = 0; v < 8; ++v) {
        const int k = wmma_k_base(v, lane);
        f[2 * v]     = mrow[(size_t)n * ldm + k];
        f[2 * v + 1] = mrow[(size_t)n * ldm + k + 1];
    }
    return f;
}

// ---------------------------------------------------------------------------
// Tensor Data Mover: 2-D bf16 tile (tile_d1 rows x tile_d0 elems, row stride
// stride0 elems) from global to contiguous LDS. D# per ISA ch.8.
// ---------------------------------------------------------------------------
__device__ __forceinline__ void tdm_load_2d_bf16(const void* gaddr, unsigned lds_addr,
                                                 unsigned tile_d0, unsigned tile_d1,
                                                 unsigned stride0) {
    const unsigned long long ga = (unsigned long long)gaddr;
    v4u g0;
    g0[0] = 1u;                                           // count=1, user mode
    g0[1] = lds_addr;                                     // LDS byte address
    g0[2] = (unsigned)(ga & 0xFFFFFFFFu);                 // global_addr[31:0]
    g0[3] = (unsigned)((ga >> 32) & 0x01FFFFFFu) | (2u << 30); // addr[56:32]|type=2

    const unsigned td0 = tile_d0, td1 = tile_d1;          // tensor dims = tile dims
    v8i_ g1;
    g1[0] = (int)(1u << 16);                              // data_size=1 (2 bytes)
    g1[1] = (int)((td0 & 0xFFFFu) << 16);                 // tensor_dim0[15:0]
    g1[2] = (int)((td0 >> 16) | ((td1 & 0xFFFFu) << 16)); // dim0[31:16]|dim1[15:0]
    g1[3] = (int)((td1 >> 16) | (tile_d0 << 16));         // dim1[31:16]|tile_dim0
    g1[4] = (int)(tile_d1 & 0xFFFFu);                     // tile_dim1, tile_dim2=0
    g1[5] = (int)stride0;                                 // tensor_dim0_stride[31:0]
    g1[6] = 0;
    g1[7] = 0;

    v4i_ z4 = {0, 0, 0, 0};
#if __clang_major__ >= 23
    v8i_ z8 = {0, 0, 0, 0, 0, 0, 0, 0};
    __builtin_amdgcn_tensor_load_to_lds(g0, g1, z4, z4, z8, 0);
#else
    __builtin_amdgcn_tensor_load_to_lds(g0, g1, z4, z4, 0);
#endif
}

__device__ __forceinline__ unsigned lds_off(const void* p) {
    return (unsigned)(unsigned long long)(uintptr_t)p;    // low 32 bits = LDS offset
}

// ---------------------------------------------------------------------------
// Prep kernels: f32 -> bf16 convert, and transpose+convert for weights.
// ---------------------------------------------------------------------------
__global__ void __launch_bounds__(256)
cvt_bf16_kernel(const float* __restrict__ x, __bf16* __restrict__ y, int n4) {
    const int i = blockIdx.x * 256 + threadIdx.x;
    if (i < n4) {
        const float4 f = ((const float4*)x)[i];
        __bf16* o = y + (size_t)i * 4;
        o[0] = (__bf16)f.x; o[1] = (__bf16)f.y; o[2] = (__bf16)f.z; o[3] = (__bf16)f.w;
    }
}

// W[K,N] f32 -> WT[N,K] bf16 (32x32 tiles, 256 threads = 32x8)
__global__ void __launch_bounds__(256)
transpose_cvt_kernel(const float* __restrict__ W, __bf16* __restrict__ WT,
                     int Kd, int Nd) {
    __shared__ float tile[32][33];
    const int tx = threadIdx.x & 31;
    const int ty = threadIdx.x >> 5;
    const int bx = blockIdx.x % (Nd / 32);
    const int by = blockIdx.x / (Nd / 32);
#pragma unroll
    for (int i = 0; i < 32; i += 8)
        tile[ty + i][tx] = W[(size_t)(by * 32 + ty + i) * Nd + bx * 32 + tx];
    __syncthreads();
#pragma unroll
    for (int i = 0; i < 32; i += 8)
        WT[(size_t)(bx * 32 + ty + i) * Kd + by * 32 + tx] = (__bf16)tile[tx][ty + i];
}

// ---------------------------------------------------------------------------
// GEMM + bias: C[M,N] = A[M,K](bf16) * WT[N,K]^T(bf16) + bias[N]
// 8 waves/WG, WG tile 128x64, wave tile 16x64. W tiles (64 rows x 32 k) are
// TDM-staged into double-buffered LDS, pipelined against compute.
// ---------------------------------------------------------------------------
template <typename OT>
__global__ void __launch_bounds__(256)
gemm_bias_kernel(const __bf16* __restrict__ A, const __bf16* __restrict__ WT,
                 const float* __restrict__ bias, OT* __restrict__ C,
                 int M, int N, int K) {
    __shared__ __bf16 lds_w[2][64 * 32];                  // 2 x 4 KB W tiles

    const int lane = threadIdx.x & 31;
    const int wave = threadIdx.x >> 5;
    const int tilesN = N / 64;
    const int r0 = (blockIdx.x / tilesN) * 128 + wave * 16;
    const int c0 = (blockIdx.x % tilesN) * 64;

    v8f acc[4];
#pragma unroll
    for (int t = 0; t < 4; ++t) acc[t] = vzero();

    const __bf16* arow = A + (size_t)r0 * K;
    const __bf16* wrow = WT + (size_t)c0 * K;
    const int nsteps = K / 32;

    if (wave == 0)                                        // prologue: tile 0
        tdm_load_2d_bf16(wrow, lds_off(lds_w[0]), 32, 64, (unsigned)K);

    for (int i = 0; i < nsteps; ++i) {
        const int cur = i & 1;
        if (wave == 0) {
            if (i + 1 < nsteps) {                         // stream tile i+1
                tdm_load_2d_bf16(wrow + (size_t)(i + 1) * 32,
                                 lds_off(lds_w[cur ^ 1]), 32, 64, (unsigned)K);
                __builtin_amdgcn_s_wait_tensorcnt(1);     // tile i resident
            } else {
                __builtin_amdgcn_s_wait_tensorcnt(0);
            }
        }
        __syncthreads();

        const v16bf a = load_a_frag(arow + i * 32, K, lane);
        v16bf bw[4];
#pragma unroll
        for (int t = 0; t < 4; ++t)
            bw[t] = load_bt_frag(lds_w[cur] + (size_t)t * 16 * 32, 32, lane);
#pragma unroll
        for (int t = 0; t < 4; ++t)
            acc[t] = wmma_bf16(a, bw[t], acc[t]);

        __syncthreads();                                  // release buffer `cur`
    }

    const int n  = lane & 15;
    const int mh = 8 * (lane >> 4);
#pragma unroll
    for (int t = 0; t < 4; ++t) {
        const float bv = bias[c0 + t * 16 + n];
#pragma unroll
        for (int r = 0; r < 8; ++r)
            C[(size_t)(r0 + r + mh) * N + c0 + t * 16 + n] = (OT)(acc[t][r] + bv);
    }
}

// ---------------------------------------------------------------------------
// Flash attention. q/k/v bf16 in [B, L, H*64]. 8 waves/WG sharing one (b,h);
// wave w owns 16 query rows. K/V 32-key head slices are TDM-staged into
// double-buffered LDS and pipelined against S/softmax/O compute.
// ---------------------------------------------------------------------------
__global__ void __launch_bounds__(256)
attention_kernel(const __bf16* __restrict__ q, const __bf16* __restrict__ k,
                 const __bf16* __restrict__ v, __bf16* __restrict__ ctx) {
    __shared__ __bf16 lds_k[2][32 * 64];                  // 2 x 4 KB
    __shared__ __bf16 lds_v[2][32 * 64];                  // 2 x 4 KB
    __shared__ __bf16 lds_p[8][16 * 32];                  // 8 KB P staging

    const int lane = threadIdx.x & 31;
    const int wave = threadIdx.x >> 5;
    const int gw = blockIdx.x * 8 + wave;
    const int qt = gw & (LL / 16 - 1);
    const int bh = gw >> 7;                               // same for all 8 waves
    const int h  = bh & (HH - 1);
    const int b  = bh >> 4;

    const int ld = HH * DKK;                              // 1024
    const size_t base = ((size_t)b * LL) * ld + (size_t)h * DKK;
    const __bf16* qp = q + base;
    const __bf16* kp = k + base;
    const __bf16* vp = v + base;
    const int qr0 = qt * 16;

    const v16bf aQ0 = load_a_frag(qp + (size_t)qr0 * ld, ld, lane);
    const v16bf aQ1 = load_a_frag(qp + (size_t)qr0 * ld + 32, ld, lane);

    v8f O[4];
#pragma unroll
    for (int t = 0; t < 4; ++t) O[t] = vzero();
    float rmax[8], rsum[8];
#pragma unroll
    for (int r = 0; r < 8; ++r) { rmax[r] = -3.0e38f; rsum[r] = 0.f; }

    const float scale = 0.125f;                           // 1/sqrt(64)
    const int n  = lane & 15;
    const int mh = 8 * (lane >> 4);
    __bf16* pl = lds_p[wave];
    const int nsteps = LL / 32;

    if (wave == 0) {                                      // prologue: tile 0
        tdm_load_2d_bf16(kp, lds_off(lds_k[0]), 64, 32, (unsigned)ld);
        tdm_load_2d_bf16(vp, lds_off(lds_v[0]), 64, 32, (unsigned)ld);
    }

    for (int i = 0; i < nsteps; ++i) {
        const int cur = i & 1;
        if (wave == 0) {
            if (i + 1 < nsteps) {                         // stream tile i+1 (2 ops)
                const size_t off = (size_t)(i + 1) * 32 * ld;
                tdm_load_2d_bf16(kp + off, lds_off(lds_k[cur ^ 1]), 64, 32, (unsigned)ld);
                tdm_load_2d_bf16(vp + off, lds_off(lds_v[cur ^ 1]), 64, 32, (unsigned)ld);
                __builtin_amdgcn_s_wait_tensorcnt(2);     // tile i resident
            } else {
                __builtin_amdgcn_s_wait_tensorcnt(0);
            }
        }
        __syncthreads();

        const __bf16* lk = lds_k[cur];
        const __bf16* lv = lds_v[cur];

        // ---- S = Q * K^T (two 16-col C tiles); load all frags, then WMMA ----
        v16bf bk[4];
        bk[0] = load_bt_frag(lk,                64, lane);
        bk[1] = load_bt_frag(lk + 32,           64, lane);
        bk[2] = load_bt_frag(lk + 16 * 64,      64, lane);
        bk[3] = load_bt_frag(lk + 16 * 64 + 32, 64, lane);
        v8f S0 = vzero(), S1 = vzero();
        S0 = wmma_bf16(aQ0, bk[0], S0);
        S0 = wmma_bf16(aQ1, bk[1], S0);
        S1 = wmma_bf16(aQ0, bk[2], S1);
        S1 = wmma_bf16(aQ1, bk[3], S1);

        // ---- online softmax over the 32-key block ----
#pragma unroll
        for (int r = 0; r < 8; ++r) {
            float s0 = S0[r] * scale;
            float s1 = S1[r] * scale;
            float bm = fmaxf(s0, s1);
            bm = fmaxf(bm, __shfl_xor(bm, 1, 16));
            bm = fmaxf(bm, __shfl_xor(bm, 2, 16));
            bm = fmaxf(bm, __shfl_xor(bm, 4, 16));
            bm = fmaxf(bm, __shfl_xor(bm, 8, 16));
            const float nmax  = fmaxf(rmax[r], bm);
            const float alpha = __expf(rmax[r] - nmax);
            const float p0 = __expf(s0 - nmax);
            const float p1 = __expf(s1 - nmax);
            float bs = p0 + p1;
            bs += __shfl_xor(bs, 1, 16);
            bs += __shfl_xor(bs, 2, 16);
            bs += __shfl_xor(bs, 4, 16);
            bs += __shfl_xor(bs, 8, 16);
            rsum[r] = rsum[r] * alpha + bs;
            rmax[r] = nmax;
            S0[r] = p0;
            S1[r] = p1;
#pragma unroll
            for (int t = 0; t < 4; ++t) O[t][r] *= alpha;
        }

        // ---- P: C layout -> A layout via wave-private LDS tile ----
#pragma unroll
        for (int r = 0; r < 8; ++r) {
            pl[(r + mh) * 32 + n]      = (__bf16)S0[r];
            pl[(r + mh) * 32 + 16 + n] = (__bf16)S1[r];
        }
        asm volatile("s_wait_dscnt 0x0" ::: "memory");
        const v16bf aP = load_a_frag(pl, 32, lane);

        // ---- O += P * V; load all frags, then WMMA ----
        v16bf bv[4];
#pragma unroll
        for (int t = 0; t < 4; ++t)
            bv[t] = load_b_frag(lv + t * 16, 64, lane);
#pragma unroll
        for (int t = 0; t < 4; ++t)
            O[t] = wmma_bf16(aP, bv[t], O[t]);

        __syncthreads();                                  // release buffer `cur`
    }

    __bf16* cp = ctx + ((size_t)b * LL + qr0) * ld + (size_t)h * DKK;
#pragma unroll
    for (int t = 0; t < 4; ++t) {
#pragma unroll
        for (int r = 0; r < 8; ++r)
            cp[(size_t)(r + mh) * ld + t * 16 + n] = (__bf16)(O[t][r] / rsum[r]);
    }
}

// ---------------------------------------------------------------------------
// Launch: convert activations, transpose+convert weights, 3 projection GEMMs,
// attention, output GEMM. Workspace: 120 MB bf16.
// ---------------------------------------------------------------------------
extern "C" void kernel_launch(void* const* d_in, const int* in_sizes, int n_in,
                              void* d_out, int out_size, void* d_ws, size_t ws_size,
                              hipStream_t stream) {
    const float* query = (const float*)d_in[0];
    const float* key_i = (const float*)d_in[1];
    const float* value = (const float*)d_in[2];
    const float* W_q = (const float*)d_in[3];
    const float* b_q = (const float*)d_in[4];
    const float* W_k = (const float*)d_in[5];
    const float* b_k = (const float*)d_in[6];
    const float* W_v = (const float*)d_in[7];
    const float* b_v = (const float*)d_in[8];
    const float* W_o = (const float*)d_in[9];
    const float* b_o = (const float*)d_in[10];

    const size_t act = (size_t)BB * LL * DD;              // 8M elements
    const size_t wel = (size_t)DD * DD;                   // 1M elements
    __bf16* wq   = (__bf16*)d_ws;
    __bf16* wk   = wq   + act;
    __bf16* wv   = wk   + act;
    __bf16* wctx = wv   + act;
    __bf16* xq   = wctx + act;
    __bf16* xk   = xq   + act;
    __bf16* xv   = xk   + act;
    __bf16* wtq  = xv   + act;
    __bf16* wtk  = wtq  + wel;
    __bf16* wtv  = wtk  + wel;
    __bf16* wto  = wtv  + wel;

    const int M = BB * LL;                                // 8192
    const dim3 blk(256);
    const int cvt_grid  = (int)(act / 4 / 256);           // 8192
    const int tr_grid   = (DD / 32) * (DD / 32);          // 1024
    const int gemm_grid = (M / 128) * (DD / 64);          // 1024
    const int attn_grid = BB * HH * (LL / 16) / 8;        // 1024

    cvt_bf16_kernel<<<cvt_grid, blk, 0, stream>>>(query, xq, (int)(act / 4));
    cvt_bf16_kernel<<<cvt_grid, blk, 0, stream>>>(key_i, xk, (int)(act / 4));
    cvt_bf16_kernel<<<cvt_grid, blk, 0, stream>>>(value, xv, (int)(act / 4));

    transpose_cvt_kernel<<<tr_grid, blk, 0, stream>>>(W_q, wtq, DD, DD);
    transpose_cvt_kernel<<<tr_grid, blk, 0, stream>>>(W_k, wtk, DD, DD);
    transpose_cvt_kernel<<<tr_grid, blk, 0, stream>>>(W_v, wtv, DD, DD);
    transpose_cvt_kernel<<<tr_grid, blk, 0, stream>>>(W_o, wto, DD, DD);

    gemm_bias_kernel<__bf16><<<gemm_grid, blk, 0, stream>>>(xq, wtq, b_q, wq, M, DD, DD);
    gemm_bias_kernel<__bf16><<<gemm_grid, blk, 0, stream>>>(xk, wtk, b_k, wk, M, DD, DD);
    gemm_bias_kernel<__bf16><<<gemm_grid, blk, 0, stream>>>(xv, wtv, b_v, wv, M, DD, DD);

    attention_kernel<<<attn_grid, blk, 0, stream>>>(wq, wk, wv, wctx);

    gemm_bias_kernel<float><<<gemm_grid, blk, 0, stream>>>(wctx, wto, b_o, (float*)d_out, M, DD, DD);
}
